// LSTM_5446018531658
// MI455X (gfx1250) — matrix operand order repaired and checked by reading, not compiled, not run on hardware
//
#include <hip/hip_runtime.h>

typedef __attribute__((ext_vector_type(16))) _Float16 v16h;
typedef __attribute__((ext_vector_type(8)))  _Float16 v8h;
typedef __attribute__((ext_vector_type(8)))  float    v8f;

#define BATCH 8192
#define SEQ   1024
#define INF   5
#define HID   64
#define OUTN  7
#define NG    256           // 4*HID
#define NT    16            // N tiles of 16 over NG (gate-interleaved order)
#define WAVES 4
#define TPB   (WAVES * 32)

// LDS layout (bytes)
#define OFF_WIH1  0                         // 16*1*32*16 halfs = 16384 B
#define OFF_WHH1  (OFF_WIH1 + 16384)        // 16*2*32*16 halfs = 32768 B
#define OFF_WIH2  (OFF_WHH1 + 32768)
#define OFF_WHH2  (OFF_WIH2 + 32768)
#define OFF_B2    (OFF_WHH2 + 32768)        // 256 f32 (gate-interleaved, layer 2 only)
#define OFF_HST   (OFF_B2 + 1024)           // 4 waves * 16*64 halfs = 8192 B
#define LDS_BYTES (OFF_HST + WAVES * HID * 16 * 2)

union frag16 { v16h f; v8h h[2]; };

#if __has_builtin(__builtin_amdgcn_tanhf)
__device__ __forceinline__ float tnh(float x) { return __builtin_amdgcn_tanhf(x); }
#else
__device__ __forceinline__ float tnh(float x) {
    return 2.0f * __builtin_amdgcn_rcpf(1.0f + __expf(-2.0f * x)) - 1.0f;
}
#endif
__device__ __forceinline__ float sigm(float x) { return 0.5f * tnh(0.5f * x) + 0.5f; }

// Stage weight matrix W [NG x Kact] (row-major) into LDS as WMMA B-fragments,
// gate-interleaved N tiles: tile nt = ht*4 + g covers original columns
// n = g*64 + ht*16 + (lane&15).  Fragment element j of lane l in chunk c holds
// W[n, 32c + (l<16 ? 0 : 8) + (j<8 ? j : j+8)], zero-padded past Kact.
__device__ inline void stageB(_Float16* dst, const float* W, int Kact, int nch, int tid) {
    int total = NT * nch * 512;
    for (int idx = tid; idx < total; idx += TPB) {
        int j  = idx & 15;
        int l  = (idx >> 4) & 31;
        int c  = (idx >> 9) % nch;
        int nt = (idx >> 9) / nch;
        int n  = (nt & 3) * HID + (nt >> 2) * 16 + (l & 15);
        int k  = 32 * c + ((l < 16) ? 0 : 8) + ((j < 8) ? j : j + 8);
        float v = (k < Kact) ? W[n * Kact + k] : 0.0f;
        dst[idx] = (_Float16)v;
    }
}

// Layer-1 input fragments: K=32 single chunk; rows 0..4 = W_ih1 columns,
// row 5 = combined bias (consumed by the constant-1 element of the x fragment).
__device__ inline void stageBx(_Float16* dst, const float* W,
                               const float* ba, const float* bb, int tid) {
    int total = NT * 512;
    for (int idx = tid; idx < total; idx += TPB) {
        int j  = idx & 15;
        int l  = (idx >> 4) & 31;
        int nt = idx >> 9;
        int n  = (nt & 3) * HID + (nt >> 2) * 16 + (l & 15);
        int k  = ((l < 16) ? 0 : 16) + j;
        float v = 0.0f;
        if (k < INF)       v = W[n * INF + k];
        else if (k == INF) v = ba[n] + bb[n];
        dst[idx] = (_Float16)v;
    }
}

extern "C" __global__ void __launch_bounds__(TPB, 1)
lstm2_wmma_kernel(const float* __restrict__ x,
                  const float* __restrict__ W_ih1, const float* __restrict__ W_hh1,
                  const float* __restrict__ b_ih1, const float* __restrict__ b_hh1,
                  const float* __restrict__ W_ih2, const float* __restrict__ W_hh2,
                  const float* __restrict__ b_ih2, const float* __restrict__ b_hh2,
                  const float* __restrict__ W_out, const float* __restrict__ b_out,
                  float* __restrict__ out)
{
    extern __shared__ char smem[];
    _Float16* wih1 = (_Float16*)(smem + OFF_WIH1);
    _Float16* whh1 = (_Float16*)(smem + OFF_WHH1);
    _Float16* wih2 = (_Float16*)(smem + OFF_WIH2);
    _Float16* whh2 = (_Float16*)(smem + OFF_WHH2);
    float*    bg2  = (float*)   (smem + OFF_B2);

    int tid  = threadIdx.x;
    int lane = tid & 31;
    int wave = tid >> 5;
    int colsel = lane & 15;

    // ---- one-time weight staging (f32 -> f16, gate-interleaved B fragments) ----
    stageBx(wih1, W_ih1, b_ih1, b_hh1, tid);   // K padded 5 -> 32, bias in row 5
    stageB (whh1, W_hh1, HID, 2, tid);
    stageB (wih2, W_ih2, HID, 2, tid);
    stageB (whh2, W_hh2, HID, 2, tid);
    for (int i = tid; i < NG; i += TPB) {
        int nt = i >> 4, cc = i & 15;
        int n = (nt & 3) * HID + (nt >> 2) * 16 + cc;
        bg2[i] = b_ih2[n] + b_hh2[n];
    }
    __syncthreads();

    // layer-2 biases resident in registers (per-lane, per gate-interleaved tile)
    float b2r[NT];
#pragma unroll
    for (int nt = 0; nt < NT; ++nt) b2r[nt] = bg2[nt * 16 + colsel];

    _Float16* hst = (_Float16*)(smem + OFF_HST) + wave * (16 * HID); // per-wave staging tile

    int b0 = (blockIdx.x * WAVES + wave) * 16;   // 16 batch rows per wave

    // State: c in C/D layout (4 hid-tiles x 8 rows/lane), h as A-fragments (2 K-chunks)
    v8f  c1[4] = {}, c2[4] = {};
    v16h h1a0 = {}, h1a1 = {}, h2a0 = {}, h2a1 = {};

    int k0sel  = (lane < 16) ? 0 : 8;    // A-fragment low-half K base within chunk
    int rowsel = (lane < 16) ? 0 : 8;    // D-layout row base

#pragma unroll 1
    for (int t = 0; t < SEQ; ++t) {
        // ---- x_t A-fragment: K slots 0..4 = x, slot 5 = 1.0 (bias row) ----
        v16h xa = {};
        if (lane < 16) {
            const float* xp = x + ((long)(b0 + lane) * SEQ + t) * INF;
#pragma unroll
            for (int i = 0; i < INF; ++i) xa[i] = (_Float16)xp[i];
            xa[INF] = (_Float16)1.0f;
            __builtin_prefetch(xp + 16 * INF, 0, 1);   // ~16 steps ahead
        }

        // ============ layer 1: gates = [x,1]@[Wih1;b]^T + h1@Whh1^T ============
#pragma unroll
        for (int ht = 0; ht < 4; ++ht) {
            v8f a[4];
#pragma unroll
            for (int g = 0; g < 4; ++g) {
                int nt = ht * 4 + g;
                v8f zero = {};
                v8f aa;
                v16h w;
                w = *(const v16h*)(wih1 + nt * 512 + lane * 16);
                aa = __builtin_amdgcn_wmma_f32_16x16x32_f16(false, xa,   false, w, (short)0, zero, false, false);
                w = *(const v16h*)(whh1 + (nt * 2 + 0) * 512 + lane * 16);
                aa = __builtin_amdgcn_wmma_f32_16x16x32_f16(false, h1a0, false, w, (short)0, aa, false, false);
                w = *(const v16h*)(whh1 + (nt * 2 + 1) * 512 + lane * 16);
                aa = __builtin_amdgcn_wmma_f32_16x16x32_f16(false, h1a1, false, w, (short)0, aa, false, false);
                a[g] = aa;
            }
            // activations for hid block ht (a[0..3] = i,f,g,o), stage h1 -> LDS as f16
#pragma unroll
            for (int e = 0; e < 8; ++e) {
                float iv = sigm(a[0][e]);
                float fv = sigm(a[1][e]);
                float gv = tnh (a[2][e]);
                float ov = sigm(a[3][e]);
                float cn = fv * c1[ht][e] + iv * gv;
                c1[ht][e] = cn;
                float hn = ov * tnh(cn);
                hst[(rowsel + e) * HID + ht * 16 + colsel] = (_Float16)hn;
            }
        }
        // reload h1 as A-fragments (DS ops are in-order within a wave)
        {
            frag16 f0, f1;
            const _Float16* hp = hst + colsel * HID;
            f0.h[0] = *(const v8h*)(hp + k0sel);
            f0.h[1] = *(const v8h*)(hp + k0sel + 16);
            f1.h[0] = *(const v8h*)(hp + 32 + k0sel);
            f1.h[1] = *(const v8h*)(hp + 32 + k0sel + 16);
            h1a0 = f0.f; h1a1 = f1.f;
        }

        // ============ layer 2: gates = h1@Wih2^T + h2@Whh2^T + b ============
#pragma unroll
        for (int ht = 0; ht < 4; ++ht) {
            v8f a[4];
#pragma unroll
            for (int g = 0; g < 4; ++g) {
                int nt = ht * 4 + g;
                float bb = b2r[nt];
                v8f aa;
#pragma unroll
                for (int e = 0; e < 8; ++e) aa[e] = bb;
                v16h w;
                w = *(const v16h*)(wih2 + (nt * 2 + 0) * 512 + lane * 16);
                aa = __builtin_amdgcn_wmma_f32_16x16x32_f16(false, h1a0, false, w, (short)0, aa, false, false);
                w = *(const v16h*)(wih2 + (nt * 2 + 1) * 512 + lane * 16);
                aa = __builtin_amdgcn_wmma_f32_16x16x32_f16(false, h1a1, false, w, (short)0, aa, false, false);
                w = *(const v16h*)(whh2 + (nt * 2 + 0) * 512 + lane * 16);
                aa = __builtin_amdgcn_wmma_f32_16x16x32_f16(false, h2a0, false, w, (short)0, aa, false, false);
                w = *(const v16h*)(whh2 + (nt * 2 + 1) * 512 + lane * 16);
                aa = __builtin_amdgcn_wmma_f32_16x16x32_f16(false, h2a1, false, w, (short)0, aa, false, false);
                a[g] = aa;
            }
#pragma unroll
            for (int e = 0; e < 8; ++e) {
                float iv = sigm(a[0][e]);
                float fv = sigm(a[1][e]);
                float gv = tnh (a[2][e]);
                float ov = sigm(a[3][e]);
                float cn = fv * c2[ht][e] + iv * gv;
                c2[ht][e] = cn;
                float hn = ov * tnh(cn);
                hst[(rowsel + e) * HID + ht * 16 + colsel] = (_Float16)hn;
            }
        }
        {
            frag16 f0, f1;
            const _Float16* hp = hst + colsel * HID;
            f0.h[0] = *(const v8h*)(hp + k0sel);
            f0.h[1] = *(const v8h*)(hp + k0sel + 16);
            f1.h[0] = *(const v8h*)(hp + 32 + k0sel);
            f1.h[1] = *(const v8h*)(hp + 32 + k0sel + 16);
            h2a0 = f0.f; h2a1 = f1.f;
        }
    }

    // ---- output projection: out = h2 @ W_out^T + b_out  (h2 is in hst, f16) ----
    if (lane < 16) {
        int row = lane;
#pragma unroll
        for (int o = 0; o < OUTN; ++o) {
            float s = b_out[o];
            for (int k = 0; k < HID; ++k)
                s += (float)hst[row * HID + k] * W_out[o * HID + k];
            out[(long)(b0 + row) * OUTN + o] = s;
        }
    }
}

extern "C" void kernel_launch(void* const* d_in, const int* in_sizes, int n_in,
                              void* d_out, int out_size, void* d_ws, size_t ws_size,
                              hipStream_t stream) {
    const float* x     = (const float*)d_in[0];
    const float* W_ih1 = (const float*)d_in[1];
    const float* W_hh1 = (const float*)d_in[2];
    const float* b_ih1 = (const float*)d_in[3];
    const float* b_hh1 = (const float*)d_in[4];
    const float* W_ih2 = (const float*)d_in[5];
    const float* W_hh2 = (const float*)d_in[6];
    const float* b_ih2 = (const float*)d_in[7];
    const float* b_hh2 = (const float*)d_in[8];
    const float* W_out = (const float*)d_in[9];
    const float* b_out = (const float*)d_in[10];
    float* out = (float*)d_out;

    (void)hipFuncSetAttribute((const void*)lstm2_wmma_kernel,
                              hipFuncAttributeMaxDynamicSharedMemorySize,
                              (int)LDS_BYTES);

    dim3 grid(BATCH / (16 * WAVES));   // 128 blocks, 4 waves each -> 512 batch tiles
    dim3 block(TPB);
    lstm2_wmma_kernel<<<grid, block, LDS_BYTES, stream>>>(
        x, W_ih1, W_hh1, b_ih1, b_hh1, W_ih2, W_hh2, b_ih2, b_hh2, W_out, b_out, out);
}
